// MIRACLE_2611340116165
// MI455X (gfx1250) — matrix operand order repaired and checked by reading, not compiled, not run on hardware
//
#include <hip/hip_runtime.h>
#include <hip/hip_bf16.h>

// Problem constants (from reference): N=32768 rows, D=160 in-dim, F=128 nets, H=32 hidden.
#define N_ROWS 32768
#define DIM    160
#define NF     128
#define NH     32

typedef __attribute__((ext_vector_type(16))) __bf16 v16bf;
typedef __attribute__((ext_vector_type(8)))  __bf16 v8bf;
typedef __attribute__((ext_vector_type(4)))  __bf16 v4bf;
typedef __attribute__((ext_vector_type(8)))  float  v8f;
typedef __attribute__((ext_vector_type(4)))  unsigned int v4u;
typedef __attribute__((ext_vector_type(8)))  int    v8i;
typedef __attribute__((ext_vector_type(4)))  int    v4i;

union BF16x16 { v16bf v; v8bf h[2]; };

// ---- workspace layout (bf16 pre-converted / pre-swizzled operands) ----
#define XBF_OFF   ((size_t)0)
#define XBF_BYTES ((size_t)N_ROWS * DIM * 2)          // 10,485,760
#define W1T_OFF   (XBF_OFF + XBF_BYTES)
#define W1T_BYTES ((size_t)NF * NH * DIM * 2)         //  1,310,720  ([f][h][d])
#define W2T_OFF   (W1T_OFF + W1T_BYTES)
#define W2T_BYTES ((size_t)NF * NH * NH * 2)          //    262,144  ([f][k_out][h_in])
#define WS_NEEDED (W2T_OFF + W2T_BYTES)

__device__ __forceinline__ float elu_f(float x) {
    return x > 0.0f ? x : (__expf(x) - 1.0f);
}

__device__ __forceinline__ v8f wmma_bf16(v16bf a, v16bf b, v8f c) {
    return __builtin_amdgcn_wmma_f32_16x16x32_bf16(
        false, a, false, b, (short)0, c, false, false);
}

// Low 32 bits of a flat shared-aperture address are the LDS byte address
// (flat->LDS mapping truncates the upper bits, ISA 10.2).
__device__ __forceinline__ unsigned lds_addr_of(const void* p) {
    return (unsigned)(unsigned long long)p;
}

// Issue one TDM 2-D tile load (tile == whole tensor), element size 2 bytes.
// D# fields per ISA 8.3/8.4: group0 = {flags, lds_addr, global_addr, type=2},
// group1 = {data_size, tensor dims, tile dims, dim0 stride}; groups 2-3 zero (2-D).
// This toolchain declares the 6-arg builtin: (v4u, v8i, v4i, v4i, v8i, i32 cpol).
__device__ __forceinline__ void tdm_load_2d_bf16(unsigned lds_byte_addr,
                                                 const void* gptr,
                                                 unsigned dim0, unsigned dim1) {
    unsigned long long ga = (unsigned long long)gptr;
    v4u g0;
    g0[0] = 1u;                                           // count=1, user descriptor
    g0[1] = lds_byte_addr;                                // lds_addr
    g0[2] = (unsigned)(ga & 0xFFFFFFFFu);                 // global_addr[31:0]
    g0[3] = (unsigned)((ga >> 32) & 0x01FFFFFFu) | (2u << 30); // addr[56:32] | type=2
    v8i g1;
    g1[0] = (int)(1u << 16);                              // data_size=1 -> 2 bytes
    g1[1] = (int)(dim0 << 16);                            // tensor_dim0[15:0]
    g1[2] = (int)((dim0 >> 16) | (dim1 << 16));           // tensor_dim0[31:16] | tensor_dim1[15:0]
    g1[3] = (int)((dim1 >> 16) | (dim0 << 16));           // tensor_dim1[31:16] | tile_dim0
    g1[4] = (int)(dim1 & 0xFFFFu);                        // tile_dim1 (tile_dim2=0)
    g1[5] = (int)dim0;                                    // tensor_dim0_stride[31:0]
    g1[6] = 0;                                            // stride hi / dim1 stride lo
    g1[7] = 0;
    v4i z4 = {0, 0, 0, 0};
    v8i z8 = {0, 0, 0, 0, 0, 0, 0, 0};
    __builtin_amdgcn_tensor_load_to_lds(g0, g1, z4, z4, z8, 0);
}

// ======================= prep kernels (run once per launch) =======================
__global__ __launch_bounds__(256) void MIRACLE_prep_xbf(const float* __restrict__ X,
                                                        __bf16* __restrict__ Xb) {
    int i = (int)(blockIdx.x * 256 + threadIdx.x) * 4;
    float4 v = *(const float4*)(X + i);
    v4bf o;
    o[0] = (__bf16)v.x; o[1] = (__bf16)v.y; o[2] = (__bf16)v.z; o[3] = (__bf16)v.w;
    *(v4bf*)(Xb + i) = o;
}

__global__ __launch_bounds__(256) void MIRACLE_prep_w1t(const float* __restrict__ W1,
                                                        __bf16* __restrict__ W1T) {
    int i = (int)(blockIdx.x * 256 + threadIdx.x);       // over [f][h][d]
    int f = i / (NH * DIM);
    int rem = i - f * (NH * DIM);
    int h = rem / DIM, d = rem - h * DIM;
    W1T[i] = (__bf16)W1[(size_t)f * DIM * NH + (size_t)d * NH + h];
}

__global__ __launch_bounds__(256) void MIRACLE_prep_w2t(const float* __restrict__ W2,
                                                        __bf16* __restrict__ W2T) {
    int i = (int)(blockIdx.x * 256 + threadIdx.x);       // over [f][k][h]
    int f = i >> 10, k = (i >> 5) & 31, h = i & 31;
    W2T[i] = (__bf16)W2[(size_t)f * NH * NH + h * NH + k];
}

// ======================= main kernel: TDM staging + bf16 WMMA =======================
// One block: one network f x 256 batch rows. 8 waves; wave w owns rows [.. + w*32, +32)
// as two 16-row M-tiles sharing B-fragments (4 WMMAs per K-step).
__global__ __launch_bounds__(256) void MIRACLE_fused_mlp_tdm(
    const __bf16* __restrict__ Xb,    // [N, D] bf16
    const __bf16* __restrict__ W1T,   // [F][H][D] bf16 (pre-transposed)
    const __bf16* __restrict__ W2T,   // [F][H][H] bf16 (pre-transposed)
    const float*  __restrict__ b1, const float* __restrict__ b2,
    const float*  __restrict__ W3, const float* __restrict__ b3,
    float* __restrict__ out)          // [N, F]
{
    __shared__ __align__(32) __bf16 sW1T[NH][DIM];        // 10 KB
    __shared__ __align__(32) __bf16 sW2T[NH][NH];         //  2 KB
    __shared__ __align__(32) __bf16 sH1[8][2][16 * NH];   // 16 KB, wave-private tiles

    const int tid   = threadIdx.x;
    const int wave  = tid >> 5;
    const int lane  = tid & 31;
    const int laneM = lane & 15;
    const int grp   = lane >> 4;

    const int f      = (int)(blockIdx.x & (NF - 1));
    const int nblk   = (int)(blockIdx.x >> 7);
    const int n_base = nblk * 256 + wave * 32;

    // ---- stage pre-swizzled bf16 weights into LDS via the Tensor Data Mover ----
    if (wave == 0) {
        tdm_load_2d_bf16(lds_addr_of(&sW1T[0][0]), W1T + (size_t)f * NH * DIM, DIM, NH);
        tdm_load_2d_bf16(lds_addr_of(&sW2T[0][0]), W2T + (size_t)f * NH * NH, NH, NH);
        __builtin_amdgcn_s_wait_tensorcnt(0);
    }
    __syncthreads();

    const float bias1_0 = b1[f * NH + laneM];
    const float bias1_1 = b1[f * NH + 16 + laneM];
    const float bias2_0 = b2[f * NH + laneM];
    const float bias2_1 = b2[f * NH + 16 + laneM];
    const float w3_0    = W3[f * NH + laneM];
    const float w3_1    = W3[f * NH + 16 + laneM];
    const float bias3   = b3[f];

    // ---- Layer 1: two 16-row M-tiles, shared B-fragments ----
    v8f c[2][2] = {};                                    // [m_tile][n_tile]
    const __bf16* xr0 = Xb + (size_t)(n_base + laneM) * DIM;
    const __bf16* xr1 = Xb + (size_t)(n_base + 16 + laneM) * DIM;

#pragma unroll
    for (int k0 = 0; k0 < DIM; k0 += 32) {
        const int ka = k0 + grp * 8;                     // A: V0-3 <- K=ka.., V4-7 <- K=ka+16..
        BF16x16 a0, a1;
        a0.h[0] = *(const v8bf*)(xr0 + ka);
        a0.h[1] = *(const v8bf*)(xr0 + ka + 16);
        a1.h[0] = *(const v8bf*)(xr1 + ka);
        a1.h[1] = *(const v8bf*)(xr1 + ka + 16);

        const int kb = k0 + grp * 8;                     // B: lane=col, V0-3 <- K=kb.., V4-7 <- K=kb+16..
        BF16x16 b0, b1v;
        b0.h[0]  = *(const v8bf*)&sW1T[laneM][kb];
        b0.h[1]  = *(const v8bf*)&sW1T[laneM][kb + 16];
        b1v.h[0] = *(const v8bf*)&sW1T[16 + laneM][kb];
        b1v.h[1] = *(const v8bf*)&sW1T[16 + laneM][kb + 16];

        c[0][0] = wmma_bf16(a0.v, b0.v,  c[0][0]);
        c[0][1] = wmma_bf16(a0.v, b1v.v, c[0][1]);
        c[1][0] = wmma_bf16(a1.v, b0.v,  c[1][0]);
        c[1][1] = wmma_bf16(a1.v, b1v.v, c[1][1]);
    }

    // ---- bias + ELU, C-layout -> row-major bf16 in wave-private LDS ----
#pragma unroll
    for (int t = 0; t < 2; ++t) {
        __bf16* h1w = &sH1[wave][t][0];
#pragma unroll
        for (int v = 0; v < 8; ++v) {
            const int row = grp * 8 + v;
            h1w[row * NH + laneM]      = (__bf16)elu_f(c[t][0][v] + bias1_0);
            h1w[row * NH + 16 + laneM] = (__bf16)elu_f(c[t][1][v] + bias1_1);
        }
    }
    asm volatile("s_wait_dscnt 0" ::: "memory");          // wave-private region, no barrier

    // ---- Layer 2: single K=32 step per tile ----
    BF16x16 bb0, bb1;
    bb0.h[0] = *(const v8bf*)&sW2T[laneM][grp * 8];
    bb0.h[1] = *(const v8bf*)&sW2T[laneM][16 + grp * 8];
    bb1.h[0] = *(const v8bf*)&sW2T[16 + laneM][grp * 8];
    bb1.h[1] = *(const v8bf*)&sW2T[16 + laneM][16 + grp * 8];

    v8f d[2][2] = {};
#pragma unroll
    for (int t = 0; t < 2; ++t) {
        const __bf16* h1w = &sH1[wave][t][0];
        BF16x16 a2;
        a2.h[0] = *(const v8bf*)&h1w[laneM * NH + grp * 8];
        a2.h[1] = *(const v8bf*)&h1w[laneM * NH + 16 + grp * 8];
        d[t][0] = wmma_bf16(a2.v, bb0.v, d[t][0]);
        d[t][1] = wmma_bf16(a2.v, bb1.v, d[t][1]);
    }

    // ---- Layer 3: elu + dot(W3) + half-wave shuffle reduction ----
#pragma unroll
    for (int t = 0; t < 2; ++t) {
        float s[8];
#pragma unroll
        for (int v = 0; v < 8; ++v) {
            const float h2_0 = elu_f(d[t][0][v] + bias2_0);
            const float h2_1 = elu_f(d[t][1][v] + bias2_1);
            s[v] = h2_0 * w3_0 + h2_1 * w3_1;
        }
#pragma unroll
        for (int off = 8; off >= 1; off >>= 1) {
#pragma unroll
            for (int v = 0; v < 8; ++v)
                s[v] += __shfl_xor(s[v], off, 16);
        }
        if (laneM == 0) {
#pragma unroll
            for (int v = 0; v < 8; ++v) {
                const int row = n_base + t * 16 + grp * 8 + v;
                out[(size_t)row * NF + f] = s[v] + bias3;
            }
        }
    }
}

// ======================= fallback (self-contained) =======================
__global__ __launch_bounds__(256) void MIRACLE_fused_mlp_fb(
    const float* __restrict__ X, const float* __restrict__ W1,
    const float* __restrict__ b1, const float* __restrict__ W2,
    const float* __restrict__ b2, const float* __restrict__ W3,
    const float* __restrict__ b3, float* __restrict__ out)
{
    __shared__ __align__(32) __bf16 sW1T[NH][176];
    __shared__ __align__(32) __bf16 sW2T[NH][48];
    __shared__ __align__(32) __bf16 sH1[8][16 * NH];

    const int tid = threadIdx.x, wave = tid >> 5, lane = tid & 31;
    const int laneM = lane & 15, grp = lane >> 4;
    const int f = (int)(blockIdx.x & (NF - 1));
    const int n_base = (int)(blockIdx.x >> 7) * 128 + wave * 16;

    const float* W1f = W1 + (size_t)f * DIM * NH;
    for (int idx = tid; idx < DIM * NH; idx += 256)
        sW1T[idx & 31][idx >> 5] = (__bf16)W1f[idx];
    const float* W2f = W2 + (size_t)f * NH * NH;
    for (int idx = tid; idx < NH * NH; idx += 256)
        sW2T[idx & 31][idx >> 5] = (__bf16)W2f[idx];
    __syncthreads();

    const float bias1_0 = b1[f * NH + laneM], bias1_1 = b1[f * NH + 16 + laneM];
    const float bias2_0 = b2[f * NH + laneM], bias2_1 = b2[f * NH + 16 + laneM];
    const float w3_0 = W3[f * NH + laneM], w3_1 = W3[f * NH + 16 + laneM];
    const float bias3 = b3[f];

    v8f c0 = {}, c1 = {};
    const float* xrow = X + (size_t)(n_base + laneM) * DIM;
#pragma unroll
    for (int k0 = 0; k0 < DIM; k0 += 32) {
        const int ka = k0 + grp * 8;
        float4 f0 = *(const float4*)(xrow + ka);
        float4 f1 = *(const float4*)(xrow + ka + 4);
        float4 f2 = *(const float4*)(xrow + ka + 16);
        float4 f3 = *(const float4*)(xrow + ka + 20);
        v16bf a;
        a[0]=(__bf16)f0.x; a[1]=(__bf16)f0.y; a[2]=(__bf16)f0.z; a[3]=(__bf16)f0.w;
        a[4]=(__bf16)f1.x; a[5]=(__bf16)f1.y; a[6]=(__bf16)f1.z; a[7]=(__bf16)f1.w;
        a[8]=(__bf16)f2.x; a[9]=(__bf16)f2.y; a[10]=(__bf16)f2.z; a[11]=(__bf16)f2.w;
        a[12]=(__bf16)f3.x; a[13]=(__bf16)f3.y; a[14]=(__bf16)f3.z; a[15]=(__bf16)f3.w;
        const int kb = k0 + grp * 8;
        BF16x16 b0, b1v;
        b0.h[0]  = *(const v8bf*)&sW1T[laneM][kb];
        b0.h[1]  = *(const v8bf*)&sW1T[laneM][kb + 16];
        b1v.h[0] = *(const v8bf*)&sW1T[16 + laneM][kb];
        b1v.h[1] = *(const v8bf*)&sW1T[16 + laneM][kb + 16];
        c0 = wmma_bf16(a, b0.v, c0);
        c1 = wmma_bf16(a, b1v.v, c1);
    }
    __bf16* h1w = &sH1[wave][0];
#pragma unroll
    for (int v = 0; v < 8; ++v) {
        const int row = grp * 8 + v;
        h1w[row * NH + laneM]      = (__bf16)elu_f(c0[v] + bias1_0);
        h1w[row * NH + 16 + laneM] = (__bf16)elu_f(c1[v] + bias1_1);
    }
    asm volatile("s_wait_dscnt 0" ::: "memory");
    BF16x16 a2, bb0, bb1;
    a2.h[0] = *(const v8bf*)&h1w[laneM * NH + grp * 8];
    a2.h[1] = *(const v8bf*)&h1w[laneM * NH + 16 + grp * 8];
    bb0.h[0] = *(const v8bf*)&sW2T[laneM][grp * 8];
    bb0.h[1] = *(const v8bf*)&sW2T[laneM][16 + grp * 8];
    bb1.h[0] = *(const v8bf*)&sW2T[16 + laneM][grp * 8];
    bb1.h[1] = *(const v8bf*)&sW2T[16 + laneM][16 + grp * 8];
    v8f d0 = {}, d1 = {};
    d0 = wmma_bf16(a2.v, bb0.v, d0);
    d1 = wmma_bf16(a2.v, bb1.v, d1);
    float s[8];
#pragma unroll
    for (int v = 0; v < 8; ++v) {
        const float h2_0 = elu_f(d0[v] + bias2_0);
        const float h2_1 = elu_f(d1[v] + bias2_1);
        s[v] = h2_0 * w3_0 + h2_1 * w3_1;
    }
#pragma unroll
    for (int off = 8; off >= 1; off >>= 1)
#pragma unroll
        for (int v = 0; v < 8; ++v) s[v] += __shfl_xor(s[v], off, 16);
    if (laneM == 0)
#pragma unroll
        for (int v = 0; v < 8; ++v)
            out[(size_t)(n_base + grp * 8 + v) * NF + f] = s[v] + bias3;
}

extern "C" void kernel_launch(void* const* d_in, const int* in_sizes, int n_in,
                              void* d_out, int out_size, void* d_ws, size_t ws_size,
                              hipStream_t stream) {
    (void)in_sizes; (void)n_in; (void)out_size;
    const float* X  = (const float*)d_in[0];
    const float* W1 = (const float*)d_in[1];
    const float* b1 = (const float*)d_in[2];
    const float* W2 = (const float*)d_in[3];
    const float* b2 = (const float*)d_in[4];
    const float* W3 = (const float*)d_in[5];
    const float* b3 = (const float*)d_in[6];
    float* out = (float*)d_out;

    if (ws_size >= WS_NEEDED && d_ws != nullptr) {
        char* ws = (char*)d_ws;
        __bf16* Xb  = (__bf16*)(ws + XBF_OFF);
        __bf16* W1T = (__bf16*)(ws + W1T_OFF);
        __bf16* W2T = (__bf16*)(ws + W2T_OFF);
        hipLaunchKernelGGL(MIRACLE_prep_xbf, dim3(N_ROWS * DIM / (256 * 4)), dim3(256), 0, stream, X, Xb);
        hipLaunchKernelGGL(MIRACLE_prep_w1t, dim3(NF * NH * DIM / 256), dim3(256), 0, stream, W1, W1T);
        hipLaunchKernelGGL(MIRACLE_prep_w2t, dim3(NF * NH * NH / 256), dim3(256), 0, stream, W2, W2T);
        // 128 consecutive blocks sweep all nets over one L2-resident 256-row X tile.
        hipLaunchKernelGGL(MIRACLE_fused_mlp_tdm, dim3(NF * (N_ROWS / 256)), dim3(256), 0, stream,
                           Xb, W1T, W2T, b1, b2, W3, b3, out);
    } else {
        hipLaunchKernelGGL(MIRACLE_fused_mlp_fb, dim3(NF * (N_ROWS / 128)), dim3(256), 0, stream,
                           X, W1, b1, W2, b2, W3, b3, out);
    }
}
